// PegasusXGlobalLocalAttention_4509715661041
// MI455X (gfx1250) — matrix operand order repaired and checked by reading, not compiled, not run on hardware
//
#include <hip/hip_runtime.h>
#include <hip/hip_bf16.h>

#define E_DIM 1024
#define NHEAD 16
#define DHEAD 64
#define BSZ   512
#define GLEN  128
#define SEQ   8192

typedef unsigned int   u32;
typedef unsigned short u16;
typedef __attribute__((ext_vector_type(4)))  u32    u32x4;
typedef __attribute__((ext_vector_type(8)))  float  v8f;
typedef __attribute__((ext_vector_type(16))) __bf16 v16bf;

union V16 { v16bf v; u32x4 q[2]; u32 w[8]; u16 h[16]; };

__device__ __forceinline__ u16 f2bf(float f) {
  u32 u = __float_as_uint(f);
  u32 r = u + 0x7fffu + ((u >> 16) & 1u);   // round-to-nearest-even
  return (u16)(r >> 16);
}

__device__ __forceinline__ v8f wmma_bf16(v16bf a, v16bf b, v8f c) {
  return __builtin_amdgcn_wmma_f32_16x16x32_bf16(false, a, false, b, (short)0, c,
                                                 false, false);
}

// ---------------------------------------------------------------- elementwise
__global__ __launch_bounds__(256) void cvt_bf16_kernel(
    const float* __restrict__ in, u16* __restrict__ out, int n) {
  int i = blockIdx.x * 256 + threadIdx.x;
  if (i < n) out[i] = f2bf(in[i]);
}

// W [E x E] f32 row-major  ->  WT [E x E] bf16 with WT[n][k] = W[k][n]
__global__ __launch_bounds__(256) void transpose_bf16_kernel(
    const float* __restrict__ in, u16* __restrict__ out) {
  __shared__ float tile[32][33];
  int x  = blockIdx.x * 32 + threadIdx.x;
  int y0 = blockIdx.y * 32;
  for (int jj = 0; jj < 4; ++jj) {
    int j = threadIdx.y + jj * 8;
    tile[j][threadIdx.x] = in[(size_t)(y0 + j) * E_DIM + x];
  }
  __syncthreads();
  int ox  = y0 + threadIdx.x;
  int oy0 = blockIdx.x * 32;
  for (int jj = 0; jj < 4; ++jj) {
    int j = threadIdx.y + jj * 8;
    out[(size_t)(oy0 + j) * E_DIM + ox] = f2bf(tile[threadIdx.x][j]);
  }
}

// ---------------------------------------------------------------- WMMA GEMM
// C = alpha * A @ B  with A [M x K] bf16 row-major, BT [N x K] bf16 row-major
// (B pre-transposed).  Block tile (WRF*64) x 128, 8 waves (4x2) each owning a
// (WRF*16) x 64 tile -> WRF*4 WMMAs per K-step of 32.
//   WRF=4: 256x128 block tile, 16 WMMAs / 16 ds_load_b128 per wave per step.
//   WRF=2: 128x128 block tile (for M=128 matrices).
template <bool F32OUT, int WRF>
__global__ __launch_bounds__(256) void gemm_wmma_kernel(
    const u16* __restrict__ A, const u16* __restrict__ BT,
    void* __restrict__ Cout, int M, int N, int K, float alpha) {
  const int BM = WRF * 64;
  __shared__ u16 As[WRF * 64][40];
  __shared__ u16 Bs[128][40];

  const int m0 = blockIdx.y * BM, n0 = blockIdx.x * 128;
  const int w = threadIdx.x >> 5, lane = threadIdx.x & 31;
  const int hf = lane >> 4, ln = lane & 15;
  const int wr = w >> 1, wc = w & 1;          // 4x2 wave grid

  v8f acc[WRF][4];
#pragma unroll
  for (int i = 0; i < WRF; ++i)
#pragma unroll
    for (int j = 0; j < 4; ++j)
#pragma unroll
      for (int e = 0; e < 8; ++e) acc[i][j][e] = 0.f;

  for (int k0 = 0; k0 < K; k0 += 32) {
    __syncthreads();
    const int rr = threadIdx.x >> 1, c0 = (threadIdx.x & 1) * 16;
#pragma unroll
    for (int i = 0; i < BM / 128; ++i) {
      const int r = i * 128 + rr;
      const u16* ap = A + (size_t)(m0 + r) * K + k0 + c0;
      *(u32x4*)&As[r][c0]     = *(const u32x4*)ap;
      *(u32x4*)&As[r][c0 + 8] = *(const u32x4*)(ap + 8);
      if (k0 + 32 < K) __builtin_prefetch(ap + 32, 0, 1);
    }
    {
      const u16* bp = BT + (size_t)(n0 + rr) * K + k0 + c0;
      *(u32x4*)&Bs[rr][c0]     = *(const u32x4*)bp;
      *(u32x4*)&Bs[rr][c0 + 8] = *(const u32x4*)(bp + 8);
      if (k0 + 32 < K) __builtin_prefetch(bp + 32, 0, 1);
    }
    __syncthreads();

    V16 af[WRF], bf[4];
#pragma unroll
    for (int rf = 0; rf < WRF; ++rf) {
      const u16* p = &As[wr * (WRF * 16) + rf * 16 + ln][0];
      af[rf].q[0] = *(const u32x4*)(p + hf * 8);
      af[rf].q[1] = *(const u32x4*)(p + 16 + hf * 8);
    }
#pragma unroll
    for (int cf = 0; cf < 4; ++cf) {
      const u16* p = &Bs[wc * 64 + cf * 16 + ln][hf * 8];
      bf[cf].q[0] = *(const u32x4*)p;
      bf[cf].q[1] = *(const u32x4*)(p + 16);
    }
#pragma unroll
    for (int rf = 0; rf < WRF; ++rf)
#pragma unroll
      for (int cf = 0; cf < 4; ++cf)
        acc[rf][cf] = wmma_bf16(af[rf].v, bf[cf].v, acc[rf][cf]);
  }

#pragma unroll
  for (int rf = 0; rf < WRF; ++rf)
#pragma unroll
    for (int cf = 0; cf < 4; ++cf)
#pragma unroll
      for (int e = 0; e < 8; ++e) {
        int row = m0 + wr * (WRF * 16) + rf * 16 + e + 8 * hf;
        int col = n0 + wc * 64 + cf * 16 + ln;
        float vv = acc[rf][cf][e] * alpha;
        if constexpr (F32OUT)
          ((float*)Cout)[(size_t)row * N + col] = vv;
        else
          ((u16*)Cout)[(size_t)row * N + col] = f2bf(vv);
      }
}

// ------------------------------------------------------- flash attention WMMA
// Keys = [gTiles(=2) tiles of global K/V] ++ [lTiles tiles of local K/V
// starting at (qBase & blockMask)].  Additive mask applies to local keys.
// Each wave owns RF*16 query rows; online softmax; ctx written bf16 [L,E].
template <int RF>
__global__ __launch_bounds__(256) void flash_attn_kernel(
    const u16* __restrict__ Q,
    const u16* __restrict__ Kg, const u16* __restrict__ Vg,
    const u16* __restrict__ Kl, const u16* __restrict__ Vl,
    const float* __restrict__ mask, u16* __restrict__ Ctx,
    int lTiles, int blockMask) {
  const int GT = GLEN / 64;  // 2 global key tiles
  const int h  = blockIdx.y;
  const int qb = blockIdx.x * (8 * RF * 16);
  const int w = threadIdx.x >> 5, lane = threadIdx.x & 31;
  const int hf = lane >> 4, ln = lane & 15;
  const int rowbase  = qb + w * (RF * 16);
  const int kLocBase = qb & blockMask;

  __shared__ u16 Ks[64][72];        // key tile, row-major   [key][d]
  __shared__ u16 Vs[64][72];        // value tile, transposed [d][key]
  __shared__ u16 Ps[8][32][72];     // per-wave P staging (C-layout -> A-layout)

  // preload Q fragments (A-operand layout), scaled Q already stored in memory
  V16 qf[RF][2];
#pragma unroll
  for (int rf = 0; rf < RF; ++rf)
#pragma unroll
    for (int kk = 0; kk < 2; ++kk) {
      const u16* qp = Q + (size_t)(rowbase + rf * 16 + ln) * E_DIM + h * DHEAD + kk * 32;
      qf[rf][kk].q[0] = *(const u32x4*)(qp + hf * 8);
      qf[rf][kk].q[1] = *(const u32x4*)(qp + 16 + hf * 8);
    }

  float mrow[RF][8], lrow[RF][8];
  v8f acc[RF][4];
#pragma unroll
  for (int rf = 0; rf < RF; ++rf)
#pragma unroll
    for (int e = 0; e < 8; ++e) { mrow[rf][e] = -1e30f; lrow[rf][e] = 0.f; }
#pragma unroll
  for (int rf = 0; rf < RF; ++rf)
#pragma unroll
    for (int cf = 0; cf < 4; ++cf)
#pragma unroll
      for (int e = 0; e < 8; ++e) acc[rf][cf][e] = 0.f;

  const int nT = GT + lTiles;
  for (int t = 0; t < nT; ++t) {
    const bool loc = (t >= GT);
    const u16* Ksrc = loc ? Kl : Kg;
    const u16* Vsrc = loc ? Vl : Vg;
    const int krow0 = loc ? (kLocBase + (t - GT) * 64) : (t * 64);

    __syncthreads();
    {
      const int r = threadIdx.x >> 2, c0 = (threadIdx.x & 3) * 16;
      const u16* kp = Ksrc + (size_t)(krow0 + r) * E_DIM + h * DHEAD + c0;
      *(u32x4*)&Ks[r][c0]     = *(const u32x4*)kp;
      *(u32x4*)&Ks[r][c0 + 8] = *(const u32x4*)(kp + 8);
      const u16* vp = Vsrc + (size_t)(krow0 + r) * E_DIM + h * DHEAD + c0;
      V16 vv;
      vv.q[0] = *(const u32x4*)vp;
      vv.q[1] = *(const u32x4*)(vp + 8);
#pragma unroll
      for (int j = 0; j < 16; ++j) Vs[c0 + j][r] = vv.h[j];   // transpose store
    }
    __syncthreads();

#pragma unroll
    for (int rf = 0; rf < RF; ++rf) {
      // ---- scores S = Q K^T for 16 rows x 64 keys
      v8f s[4];
#pragma unroll
      for (int kf = 0; kf < 4; ++kf) {
        v8f sv;
#pragma unroll
        for (int e = 0; e < 8; ++e) sv[e] = 0.f;
#pragma unroll
        for (int kk = 0; kk < 2; ++kk) {
          V16 b;
          const u16* bp = &Ks[kf * 16 + ln][kk * 32 + hf * 8];
          b.q[0] = *(const u32x4*)bp;
          b.q[1] = *(const u32x4*)(bp + 16);
          sv = wmma_bf16(qf[rf][kk].v, b.v, sv);
        }
        if (loc) {
          float mv = mask[krow0 + kf * 16 + ln];
#pragma unroll
          for (int e = 0; e < 8; ++e) sv[e] += mv;
        }
        s[kf] = sv;
      }
      // ---- online softmax (rows live in 16-lane halves)
      float tm[8];
#pragma unroll
      for (int e = 0; e < 8; ++e)
        tm[e] = fmaxf(fmaxf(s[0][e], s[1][e]), fmaxf(s[2][e], s[3][e]));
#pragma unroll
      for (int e = 0; e < 8; ++e)
        for (int off = 1; off < 16; off <<= 1)
          tm[e] = fmaxf(tm[e], __shfl_xor(tm[e], off, 16));
      float nm[8], al[8], rs[8];
#pragma unroll
      for (int e = 0; e < 8; ++e) {
        nm[e] = fmaxf(mrow[rf][e], tm[e]);
        al[e] = __expf(mrow[rf][e] - nm[e]);
        mrow[rf][e] = nm[e];
        rs[e] = 0.f;
      }
#pragma unroll
      for (int cf = 0; cf < 4; ++cf)
#pragma unroll
        for (int e = 0; e < 8; ++e) acc[rf][cf][e] *= al[e];
#pragma unroll
      for (int kf = 0; kf < 4; ++kf)
#pragma unroll
        for (int e = 0; e < 8; ++e) {
          float p = __expf(s[kf][e] - nm[e]);
          s[kf][e] = p;
          rs[e] += p;
        }
#pragma unroll
      for (int e = 0; e < 8; ++e)
        for (int off = 1; off < 16; off <<= 1)
          rs[e] += __shfl_xor(rs[e], off, 16);
#pragma unroll
      for (int e = 0; e < 8; ++e) lrow[rf][e] = lrow[rf][e] * al[e] + rs[e];

      // ---- re-shape P (C-layout -> A-layout) through per-wave LDS
#pragma unroll
      for (int kf = 0; kf < 4; ++kf)
#pragma unroll
        for (int e = 0; e < 8; ++e)
          Ps[w][rf * 16 + e + 8 * hf][kf * 16 + ln] = f2bf(s[kf][e]);

      V16 pa[2];
#pragma unroll
      for (int kk = 0; kk < 2; ++kk) {
        const u16* pp = &Ps[w][rf * 16 + ln][kk * 32];
        pa[kk].q[0] = *(const u32x4*)(pp + hf * 8);
        pa[kk].q[1] = *(const u32x4*)(pp + 16 + hf * 8);
      }
      // ---- acc += P @ V
#pragma unroll
      for (int cf = 0; cf < 4; ++cf)
#pragma unroll
        for (int kk = 0; kk < 2; ++kk) {
          V16 bv;
          const u16* vp2 = &Vs[cf * 16 + ln][kk * 32 + hf * 8];
          bv.q[0] = *(const u32x4*)vp2;
          bv.q[1] = *(const u32x4*)(vp2 + 16);
          acc[rf][cf] = wmma_bf16(pa[kk].v, bv.v, acc[rf][cf]);
        }
    }
  }

  // ---- normalize and store context bf16 [L, E]
#pragma unroll
  for (int rf = 0; rf < RF; ++rf) {
    float inv[8];
#pragma unroll
    for (int e = 0; e < 8; ++e) inv[e] = 1.0f / lrow[rf][e];
#pragma unroll
    for (int cf = 0; cf < 4; ++cf)
#pragma unroll
      for (int e = 0; e < 8; ++e) {
        int row = rowbase + rf * 16 + e + 8 * hf;
        int col = h * DHEAD + cf * 16 + ln;
        Ctx[(size_t)row * E_DIM + col] = f2bf(acc[rf][cf][e] * inv[e]);
      }
  }
}

// ---------------------------------------------------------------- launcher
extern "C" void kernel_launch(void* const* d_in, const int* in_sizes, int n_in,
                              void* d_out, int out_size, void* d_ws, size_t ws_size,
                              hipStream_t stream) {
  const float* X    = (const float*)d_in[0];
  const float* Gx   = (const float*)d_in[1];
  const float* mask = (const float*)d_in[2];
  const float* Wq   = (const float*)d_in[3];
  const float* Wk   = (const float*)d_in[4];
  const float* Wv   = (const float*)d_in[5];
  const float* Wo   = (const float*)d_in[6];

  char* ws = (char*)d_ws;
  size_t off = 0;
  auto alloc = [&](size_t bytes) -> char* {
    char* p = ws + off;
    off = (off + bytes + 255) & ~(size_t)255;
    return p;
  };
  u16* Xb  = (u16*)alloc((size_t)SEQ * E_DIM * 2);
  u16* Gb  = (u16*)alloc((size_t)GLEN * E_DIM * 2);
  u16* WqT = (u16*)alloc((size_t)E_DIM * E_DIM * 2);
  u16* WkT = (u16*)alloc((size_t)E_DIM * E_DIM * 2);
  u16* WvT = (u16*)alloc((size_t)E_DIM * E_DIM * 2);
  u16* WoT = (u16*)alloc((size_t)E_DIM * E_DIM * 2);
  u16* Qt  = (u16*)alloc((size_t)SEQ * E_DIM * 2);
  u16* Kt  = (u16*)alloc((size_t)SEQ * E_DIM * 2);
  u16* Vt  = (u16*)alloc((size_t)SEQ * E_DIM * 2);
  u16* Qg  = (u16*)alloc((size_t)GLEN * E_DIM * 2);
  u16* Kg  = (u16*)alloc((size_t)GLEN * E_DIM * 2);
  u16* Vg  = (u16*)alloc((size_t)GLEN * E_DIM * 2);
  u16* CL  = (u16*)alloc((size_t)SEQ * E_DIM * 2);
  u16* CG  = (u16*)alloc((size_t)GLEN * E_DIM * 2);

  const float qscale = 0.125f;  // D^-0.5, D=64

  // 1) convert activations / weights(transposed) to bf16
  cvt_bf16_kernel<<<(SEQ * E_DIM) / 256, 256, 0, stream>>>(X, Xb, SEQ * E_DIM);
  cvt_bf16_kernel<<<(GLEN * E_DIM) / 256, 256, 0, stream>>>(Gx, Gb, GLEN * E_DIM);
  dim3 tt(32, 8), tg(E_DIM / 32, E_DIM / 32);
  transpose_bf16_kernel<<<tg, tt, 0, stream>>>(Wq, WqT);
  transpose_bf16_kernel<<<tg, tt, 0, stream>>>(Wk, WkT);
  transpose_bf16_kernel<<<tg, tt, 0, stream>>>(Wv, WvT);
  transpose_bf16_kernel<<<tg, tt, 0, stream>>>(Wo, WoT);

  // 2) projections (bf16 out; Q pre-scaled)
  //    big GEMMs: 256x128 block tiles (WRF=4); small M=128 GEMMs: 128x128.
  dim3 gS(E_DIM / 128, SEQ / 256), gG(E_DIM / 128, GLEN / 128);
  gemm_wmma_kernel<false, 4><<<gS, 256, 0, stream>>>(Xb, WqT, Qt, SEQ, E_DIM, E_DIM, qscale);
  gemm_wmma_kernel<false, 4><<<gS, 256, 0, stream>>>(Xb, WkT, Kt, SEQ, E_DIM, E_DIM, 1.0f);
  gemm_wmma_kernel<false, 4><<<gS, 256, 0, stream>>>(Xb, WvT, Vt, SEQ, E_DIM, E_DIM, 1.0f);
  gemm_wmma_kernel<false, 2><<<gG, 256, 0, stream>>>(Gb, WqT, Qg, GLEN, E_DIM, E_DIM, qscale);
  gemm_wmma_kernel<false, 2><<<gG, 256, 0, stream>>>(Gb, WkT, Kg, GLEN, E_DIM, E_DIM, 1.0f);
  gemm_wmma_kernel<false, 2><<<gG, 256, 0, stream>>>(Gb, WvT, Vg, GLEN, E_DIM, E_DIM, 1.0f);

  // 3) attention
  // local: keys = 128 global + 512 block-local; 32 query chunks of 256 rows
  flash_attn_kernel<2><<<dim3(SEQ / 256, NHEAD), 256, 0, stream>>>(
      Qt, Kg, Vg, Kt, Vt, mask, CL, BSZ / 64, ~(BSZ - 1));
  // global: keys = 128 global + all 8192 tokens; 128 query rows per head
  flash_attn_kernel<1><<<dim3(1, NHEAD), 256, 0, stream>>>(
      Qg, Kg, Vg, Kt, Vt, mask, CG, SEQ / 64, 0);

  // 4) output projection straight into d_out (f32): l_out then g_out
  gemm_wmma_kernel<true, 4><<<gS, 256, 0, stream>>>(CL, WoT, (float*)d_out,
                                                    SEQ, E_DIM, E_DIM, 1.0f);
  gemm_wmma_kernel<true, 2><<<gG, 256, 0, stream>>>(
      CG, WoT, (float*)d_out + (size_t)SEQ * E_DIM, GLEN, E_DIM, E_DIM, 1.0f);
}
